// _TQ4AttnModule_72550587564174
// MI455X (gfx1250) — matrix operand order repaired and checked by reading, not compiled, not run on hardware
//
#include <hip/hip_runtime.h>

typedef __attribute__((ext_vector_type(16))) _Float16 v16h;
typedef __attribute__((ext_vector_type(8)))  _Float16 v8h;
typedef __attribute__((ext_vector_type(2)))  _Float16 v2h;
typedef __attribute__((ext_vector_type(8)))  float    v8f;

#define D_DIM 128
#define S_LEN 4096
#define Q_LEN 1024
#define HQ_N  32
#define HKV_N 8
#define BQ    128
#define BS    128
#define NT    (S_LEN / BS)

__device__ __forceinline__ v8f wmma_f16(v16h a, v16h b, v8f c) {
  return __builtin_amdgcn_wmma_f32_16x16x32_f16(false, a, false, b, (short)0, c, false, false);
}

// 32-bit LDS byte offset from a generic pointer (LDS aperture keeps offset in low 32 bits)
__device__ __forceinline__ unsigned lds_off32(const void* p) {
  return (unsigned)(unsigned long)p;
}

// ---------- Phase 0: nibble -> f16 dequant with norms folded in ----------
// decK:  (HKV, S, D)  [s][d]   rows scaled by k_norm[s]
// decVT: (HKV, D, S)  [d][s]   cols scaled by v_norm[s]
__global__ __launch_bounds__(256) void tq4_dequant(
    const int* __restrict__ kp, const int* __restrict__ vp,
    const float* __restrict__ cent,
    const float* __restrict__ kn, const float* __restrict__ vn,
    _Float16* __restrict__ decK, _Float16* __restrict__ decVT)
{
  __shared__ unsigned lut[256];
  __shared__ _Float16 nk[128], nv[128];
  __shared__ __align__(16) _Float16 tVT[128 * 128];
  const int tid = threadIdx.x;
  const int kh  = blockIdx.y;
  const int s0  = blockIdx.x * 128;
  {
    _Float16 ch = (_Float16)cent[tid >> 4];   // hi nibble -> even d
    _Float16 cl = (_Float16)cent[tid & 15];   // lo nibble -> odd d
    lut[tid] = ((unsigned)__builtin_bit_cast(unsigned short, cl) << 16)
             |  (unsigned)__builtin_bit_cast(unsigned short, ch);
  }
  if (tid < 128) nk[tid]       = (_Float16)kn[(long)kh * S_LEN + s0 + tid];
  else           nv[tid - 128] = (_Float16)vn[(long)kh * S_LEN + s0 + (tid - 128)];
  __syncthreads();
  const long base = ((long)kh * S_LEN + s0) * (D_DIM / 2);
#pragma unroll 4
  for (int j = 0; j < 32; ++j) {
    int idx  = tid + j * 256;       // (srow, w)
    int srow = idx >> 6, w = idx & 63;
    v2h dk = __builtin_bit_cast(v2h, lut[kp[base + idx] & 255]) * nk[srow];
    *(unsigned*)(decK + ((long)kh * S_LEN + s0 + srow) * D_DIM + 2 * w) =
        __builtin_bit_cast(unsigned, dk);
    v2h dv = __builtin_bit_cast(v2h, lut[vp[base + idx] & 255]) * nv[srow];
    tVT[(2 * w)     * 128 + srow] = dv.x;
    tVT[(2 * w + 1) * 128 + srow] = dv.y;
  }
  __syncthreads();
  // coalesced transposed write-back
  const int d = tid >> 1, part = (tid & 1) * 64;
  const _Float16* srcp = tVT + d * 128 + part;
  _Float16* dstp = decVT + ((long)kh * D_DIM + d) * S_LEN + s0 + part;
#pragma unroll
  for (int j = 0; j < 8; ++j)
    *(v8h*)(dstp + j * 8) = *(const v8h*)(srcp + j * 8);
}

// ---------- Phase 1: q' = (Q @ R^T) * (1/sqrt(D)), f32 -> f16 ----------
__global__ __launch_bounds__(256) void tq4_qrot(const float* __restrict__ q,
                                                const float* __restrict__ rot,
                                                _Float16* __restrict__ out) {
  __shared__ float sq[16 * 128];
  const long base = (long)blockIdx.x * 16;
  for (int j = 0; j < 8; ++j) {
    int idx = threadIdx.x + j * 256;
    sq[idx] = q[base * 128 + idx];
  }
  __syncthreads();
  const int r  = threadIdx.x >> 4;
  const int d0 = (threadIdx.x & 15) * 8;
  const float SCALE = 0.08838834764831845f;  // 1/sqrt(128)
  for (int j = 0; j < 8; ++j) {
    const float* rr = rot + (long)(d0 + j) * 128;  // q'[d] = sum_e q[e]*R[d][e]
    float a = 0.f;
    for (int e = 0; e < 128; ++e) a += sq[r * 128 + e] * rr[e];
    out[(base + r) * 128 + d0 + j] = (_Float16)(a * SCALE);
  }
}

// ---------- Phase 2: flash attention, async double-buffered K/V tiles ----------
__global__ __launch_bounds__(256) void tq4_attn_main(
    const _Float16* __restrict__ qrot,   // (HQ*QL, D) f16, scale folded
    const float*    __restrict__ mask,   // (QL, S)
    const _Float16* __restrict__ decK,   // (HKV, S, D), k_norm folded
    const _Float16* __restrict__ decVT,  // (HKV, D, S), v_norm folded
    const float*    __restrict__ rot,    // (D, D)
    float*          __restrict__ out)    // (HQ, QL, D)
{
  __shared__ __align__(16) char smem[163840];
  _Float16* ldsK0  = (_Float16*)smem;               // 32 KB  [s][d] tile, buf 0
  _Float16* ldsK1  = (_Float16*)(smem + 32768);     // 32 KB  buf 1
  _Float16* ldsV0  = (_Float16*)(smem + 65536);     // 32 KB  [d][s] tile, buf 0
  _Float16* ldsV1  = (_Float16*)(smem + 98304);     // 32 KB  buf 1
  _Float16* ldsP   = (_Float16*)(smem + 131072);    // 32 KB  [wave][16][BS]
  float*    ldsO   = (float*)smem;                  // epilogue reuse: 128*129*4

  const int tid  = threadIdx.x;
  const int wave = tid >> 5, lane = tid & 31;
  const int ln   = lane & 15, hb = lane >> 4;
  const int h    = blockIdx.y;
  const int kh   = h >> 2;                          // GQA: 4 q-heads per kv-head
  const int q0   = blockIdx.x * BQ;
  const int qw   = q0 + wave * 16;

  // async-copy state: SGPR bases + 32-bit running per-thread offsets + invariant LDS dests.
  // The 24-bit instruction offset is added to BOTH addresses (ISA 08 §4.4), so chunk
  // strides are chosen equal on the LDS and global side (K: j*4096, V: j*16).
  const unsigned long gKbase = (unsigned long)(decK  + (long)kh * S_LEN * D_DIM);
  const unsigned long gVbase = (unsigned long)(decVT + (long)kh * D_DIM * S_LEN);
  unsigned voffK = (unsigned)tid * 16;                                    // += 32768/tile
  unsigned voffV = (unsigned)((tid >> 1) * 8192 + (tid & 1) * 128);       // += 256/tile
  const unsigned vdstK[2] = { lds_off32(ldsK0) + (unsigned)tid * 16,
                              lds_off32(ldsK1) + (unsigned)tid * 16 };
  const unsigned vdstV[2] = { lds_off32(ldsV0) + (unsigned)((tid >> 1) * 256 + (tid & 1) * 128),
                              lds_off32(ldsV1) + (unsigned)((tid >> 1) * 256 + (tid & 1) * 128) };

  auto issue_tile = [&](int b) {   // 8+8 x b128 per thread = one 64 KB K+V tile
    unsigned vk = vdstK[b], vv = vdstV[b];
#pragma unroll
    for (int j = 0; j < 8; ++j) {
      asm volatile("global_load_async_to_lds_b128 %0, %1, %3 offset:%c2"
                   :: "v"(vk), "v"(voffK), "i"(j * 4096), "s"(gKbase) : "memory");
      asm volatile("global_load_async_to_lds_b128 %0, %1, %3 offset:%c2"
                   :: "v"(vv), "v"(voffV), "i"(j * 16), "s"(gVbase) : "memory");
    }
    voffK += (unsigned)(BS * D_DIM * 2);   // next 32 KB K tile
    voffV += (unsigned)(BS * 2);           // next 256 B along each V row
  };

  // Q A-fragments stay in VGPRs for the whole kernel
  v16h aQ[4];
  {
    const _Float16* qptr = qrot + ((long)h * Q_LEN + qw + ln) * D_DIM;
    const int kb = hb * 16;
#pragma unroll
    for (int c = 0; c < 4; ++c) {
      v8h lo = *(const v8h*)(qptr + c * 32 + kb);
      v8h hi = *(const v8h*)(qptr + c * 32 + kb + 8);
      aQ[c] = __builtin_shufflevector(lo, hi, 0, 1, 2, 3, 4, 5, 6, 7,
                                      8, 9, 10, 11, 12, 13, 14, 15);
    }
  }

  v8f oAcc[8];
#pragma unroll
  for (int n = 0; n < 8; ++n)
#pragma unroll
    for (int r = 0; r < 8; ++r) oAcc[n][r] = 0.f;
  float mrun[8], lrun[8];
#pragma unroll
  for (int r = 0; r < 8; ++r) { mrun[r] = -1e30f; lrun[r] = 0.f; }

  _Float16* ldsPw = ldsP + wave * 16 * BS;

  issue_tile(0);

  for (int st = 0; st < NT; ++st) {
    const int s0 = st * BS;
    const _Float16* Kc = (st & 1) ? ldsK1 : ldsK0;
    const _Float16* Vc = (st & 1) ? ldsV1 : ldsV0;

    __syncthreads();                       // next buffer free of tile st-1 readers
    if (st + 1 < NT) {
      issue_tile((st + 1) & 1);
      asm volatile("s_wait_asynccnt 0x10" ::: "memory");  // tile st done (16 newer pending)
    } else {
      asm volatile("s_wait_asynccnt 0x0" ::: "memory");
    }
    __syncthreads();                       // all waves' fills of tile st visible

    // -------- scores: q'-tile (16x128) @ (C_k * k_norm)^T --------
    v8f sf[8];
#pragma unroll
    for (int n = 0; n < 8; ++n) {
      v8f accv = {};
#pragma unroll
      for (int c = 0; c < 4; ++c) {
        v16h bK = *(const v16h*)(Kc + (n * 16 + ln) * D_DIM + c * 32 + hb * 16);
        accv = wmma_f16(aQ[c], bK, accv);
      }
      sf[n] = accv;
    }

    // -------- + mask (norms already folded into decK) --------
    const float* mrow = mask + (long)(q0 + wave * 16) * S_LEN + s0;
#pragma unroll
    for (int n = 0; n < 8; ++n)
#pragma unroll
      for (int r = 0; r < 8; ++r) {
        int qrow = r + hb * 8;
        sf[n][r] += mrow[(long)qrow * S_LEN + n * 16 + ln];
      }

    // -------- online softmax (reductions within 16-lane half-groups) --------
#pragma unroll
    for (int r = 0; r < 8; ++r) {
      float tm = sf[0][r];
#pragma unroll
      for (int n = 1; n < 8; ++n) tm = fmaxf(tm, sf[n][r]);
      tm = fmaxf(tm, __shfl_xor(tm, 1, 32));
      tm = fmaxf(tm, __shfl_xor(tm, 2, 32));
      tm = fmaxf(tm, __shfl_xor(tm, 4, 32));
      tm = fmaxf(tm, __shfl_xor(tm, 8, 32));
      float nm    = fmaxf(mrun[r], tm);
      float alpha = __expf(mrun[r] - nm);
      mrun[r] = nm;
      float ps = 0.f;
      const int prow = (r + hb * 8) * BS;
#pragma unroll
      for (int n = 0; n < 8; ++n) {
        float p = __expf(sf[n][r] - nm);
        ps += p;
        ldsPw[prow + n * 16 + ln] = (_Float16)p;   // v_norm already folded into decVT
      }
      ps += __shfl_xor(ps, 1, 32);
      ps += __shfl_xor(ps, 2, 32);
      ps += __shfl_xor(ps, 4, 32);
      ps += __shfl_xor(ps, 8, 32);
      lrun[r] = lrun[r] * alpha + ps;
#pragma unroll
      for (int n = 0; n < 8; ++n) oAcc[n][r] *= alpha;
    }
    __builtin_amdgcn_wave_barrier();  // per-wave LDS ops are in-order; pin the schedule

    // -------- O += P (16x128) @ (C_v * v_norm) --------
    v16h aP[4];
    {
      const int kb = hb * 16;
      const _Float16* pptr = ldsPw + ln * BS;
#pragma unroll
      for (int c = 0; c < 4; ++c) {
        v8h lo = *(const v8h*)(pptr + c * 32 + kb);
        v8h hi = *(const v8h*)(pptr + c * 32 + kb + 8);
        aP[c] = __builtin_shufflevector(lo, hi, 0, 1, 2, 3, 4, 5, 6, 7,
                                        8, 9, 10, 11, 12, 13, 14, 15);
      }
    }
#pragma unroll
    for (int n = 0; n < 8; ++n)
#pragma unroll
      for (int c = 0; c < 4; ++c) {
        v16h bV = *(const v16h*)(Vc + (n * 16 + ln) * BS + c * 32 + hb * 16);
        oAcc[n] = wmma_f16(aP[c], bV, oAcc[n]);
      }
  }

  // -------- epilogue: normalize, stage O', apply rotation out = O' @ R --------
  __syncthreads();
#pragma unroll
  for (int n = 0; n < 8; ++n)
#pragma unroll
    for (int r = 0; r < 8; ++r) {
      int rowl = wave * 16 + r + hb * 8;
      ldsO[rowl * 129 + n * 16 + ln] = oAcc[n][r] / lrun[r];
    }
  __syncthreads();
  {
    const int i  = tid & 127;
    const int d0 = (tid >> 7) * 64;
    float acc[64];
#pragma unroll
    for (int j = 0; j < 64; ++j) acc[j] = 0.f;
    const float* orow = ldsO + i * 129;
    for (int e = 0; e < D_DIM; ++e) {
      float a = orow[e];
      const float* rr = rot + (long)e * D_DIM + d0;
#pragma unroll
      for (int j = 0; j < 64; ++j) acc[j] += a * rr[j];
    }
    float* op = out + ((long)h * Q_LEN + q0 + i) * D_DIM + d0;
#pragma unroll
    for (int j = 0; j < 64; ++j) op[j] = acc[j];
  }
}

extern "C" void kernel_launch(void* const* d_in, const int* in_sizes, int n_in,
                              void* d_out, int out_size, void* d_ws, size_t ws_size,
                              hipStream_t stream) {
  (void)in_sizes; (void)n_in; (void)out_size; (void)ws_size;
  const float* query = (const float*)d_in[0];
  const float* mask  = (const float*)d_in[1];
  const int*   kpk   = (const int*)d_in[2];
  const float* knorm = (const float*)d_in[3];
  const int*   vpk   = (const int*)d_in[4];
  const float* vnorm = (const float*)d_in[5];
  const float* cent  = (const float*)d_in[6];
  const float* rot   = (const float*)d_in[7];
  float* out = (float*)d_out;

  char* ws = (char*)d_ws;                       // 24 MB total
  _Float16* qrot  = (_Float16*)ws;              // 8 MB: (HQ*QL, D)
  _Float16* decK  = (_Float16*)(ws + (8u << 20));   // 8 MB: (HKV, S, D)
  _Float16* decVT = (_Float16*)(ws + (16u << 20));  // 8 MB: (HKV, D, S)

  tq4_dequant<<<dim3(S_LEN / 128, HKV_N), dim3(256), 0, stream>>>(
      kpk, vpk, cent, knorm, vnorm, decK, decVT);
  tq4_qrot<<<dim3(HQ_N * Q_LEN / 16), dim3(256), 0, stream>>>(query, rot, qrot);
  tq4_attn_main<<<dim3(Q_LEN / BQ, HQ_N), dim3(256), 0, stream>>>(
      qrot, mask, decK, decVT, rot, out);
}
